// RSSM_61632780697681
// MI455X (gfx1250) — matrix operand order repaired and checked by reading, not compiled, not run on hardware
//
#include <hip/hip_runtime.h>

// ---------------- problem dims ----------------
#define B_    512
#define T_    128
#define DET_  512
#define STOCH_ 32
#define EMB_  1024
#define ACT_  6
#define HID_  512
#define MIN_STD 0.1f
#define XPADK 64   // STOCH+ACT=38 padded to 64 for K alignment

typedef __attribute__((ext_vector_type(16))) __bf16 v16bf;
typedef __attribute__((ext_vector_type(8)))  float  v8f;

union Frag {
    v16bf v;
    uint4 q[2];
};

__device__ __forceinline__ v8f wmma_bf16(v16bf a, v16bf b, v8f c) {
    // 8 args: (neg_a, A, neg_b, B, c_mod, C, reuse_a, reuse_b)
    return __builtin_amdgcn_wmma_f32_16x16x32_bf16(false, a, false, b, (short)0, c, false, false);
}

// A fragment (16x32 bf16, MxK), row-major source with leading dim ld (elements).
// ISA 7.12.2: lanes 0-15 -> M=lane, K {0..7,16..23}; lanes 16-31 -> M=lane-16, K {8..15,24..31}
__device__ __forceinline__ v16bf load_a_bf16(const __bf16* A, int ld, int mbase, int kb, int lane) {
    int m   = mbase + (lane & 15);
    int klo = kb + ((lane >> 4) << 3);          // +0 or +8
    const char* p = (const char*)(A + m * ld + klo);
    Frag f;
    f.q[0] = *(const uint4*)p;                  // K = klo .. klo+7
    f.q[1] = *(const uint4*)(p + 32);           // K = klo+16 .. klo+23
    return f.v;
}

// A fragment from an fp32 source (converted in registers) — used for embedded_obs.
__device__ __forceinline__ v16bf load_a_f32(const float* A, int ld, int mbase, int kb, int lane) {
    int m   = mbase + (lane & 15);
    int klo = kb + ((lane >> 4) << 3);
    const float* p = A + m * ld + klo;
    Frag f;
#pragma unroll
    for (int i = 0; i < 8; ++i) f.v[i]     = (__bf16)p[i];
#pragma unroll
    for (int i = 0; i < 8; ++i) f.v[8 + i] = (__bf16)p[16 + i];
    return f.v;
}

// B fragment (32x16 bf16, KxN). Weights stored row-major as [N][K] (i.e. B = W^T),
// so B[k][n] = W[n*ld + k]. lanes 0-15 -> N=lane, K {0..7,16..23}; lanes 16-31 -> K {8..15,24..31}.
__device__ __forceinline__ v16bf load_b_w(const __bf16* W, int ld, int nbase, int kb, int lane) {
    int n  = nbase + (lane & 15);
    int k0 = kb + ((lane >> 4) << 3);           // +0 or +8
    const char* p = (const char*)(W + n * ld + k0);
    Frag f;
    f.q[0] = *(const uint4*)p;
    f.q[1] = *(const uint4*)(p + 32);
    return f.v;
}

__device__ __forceinline__ float sigmoidf_(float x) { return 1.0f / (1.0f + __expf(-x)); }
__device__ __forceinline__ float softplusf_(float x) {
    return (x > 0.0f) ? (x + log1pf(__expf(-x))) : log1pf(__expf(x));
}
__device__ __forceinline__ float eluf_(float x) { return (x > 0.0f) ? x : expm1f(x); }

// ---------------- prep: weights -> bf16, zero states, build x_0 ----------------
#define SZ_WIHP (3 * DET_ * XPADK)      // 98304
#define SZ_WHH  (3 * DET_ * DET_)       // 786432
#define SZ_TW1  (HID_ * DET_)           // 262144
#define SZ_TW2  (2 * STOCH_ * HID_)     // 32768
#define SZ_RW1  (HID_ * (DET_ + EMB_))  // 786432
#define SZ_RW2  (2 * STOCH_ * HID_)     // 32768
#define SZ_H    (B_ * DET_)             // 262144
#define SZ_XPAD (B_ * XPADK)            // 32768
#define SZ_TOT  (SZ_WIHP + SZ_WHH + SZ_TW1 + SZ_TW2 + SZ_RW1 + SZ_RW2 + SZ_H + SZ_H + SZ_XPAD)

__global__ void __launch_bounds__(256) k_prep(
    const float* wih, const float* whh, const float* tw1, const float* tw2,
    const float* rw1, const float* rw2, const float* act,
    __bf16* wihp, __bf16* whhb, __bf16* tw1b, __bf16* tw2b,
    __bf16* rw1b, __bf16* rw2b, __bf16* hbf0, float* hflt0, __bf16* xpad)
{
    for (int i = blockIdx.x * blockDim.x + threadIdx.x; i < SZ_TOT; i += gridDim.x * blockDim.x) {
        int idx = i;
        if (idx < SZ_WIHP) {                       // pad K: 38 -> 64
            int n = idx >> 6, k = idx & 63;
            wihp[idx] = (k < STOCH_ + ACT_) ? (__bf16)wih[n * (STOCH_ + ACT_) + k] : (__bf16)0.0f;
            continue;
        }
        idx -= SZ_WIHP;
        if (idx < SZ_WHH) { whhb[idx] = (__bf16)whh[idx]; continue; }
        idx -= SZ_WHH;
        if (idx < SZ_TW1) { tw1b[idx] = (__bf16)tw1[idx]; continue; }
        idx -= SZ_TW1;
        if (idx < SZ_TW2) { tw2b[idx] = (__bf16)tw2[idx]; continue; }
        idx -= SZ_TW2;
        if (idx < SZ_RW1) { rw1b[idx] = (__bf16)rw1[idx]; continue; }
        idx -= SZ_RW1;
        if (idx < SZ_RW2) { rw2b[idx] = (__bf16)rw2[idx]; continue; }
        idx -= SZ_RW2;
        if (idx < SZ_H)   { hbf0[idx] = (__bf16)0.0f; continue; }
        idx -= SZ_H;
        if (idx < SZ_H)   { hflt0[idx] = 0.0f; continue; }
        idx -= SZ_H;
        {   // xpad[b][64]: [ s0=0 (32) | a_0 (6) | 0 pad ]
            int m = idx >> 6, k = idx & 63;
            float v = 0.0f;
            if (k >= STOCH_ && k < STOCH_ + ACT_)
                v = act[(m * T_ + 0) * ACT_ + (k - STOCH_)];
            xpad[idx] = (__bf16)v;
        }
    }
}

// ---------------- k_gru: gates GEMMs + fused GRU cell ----------------
__device__ __forceinline__ void gru_epilogue(
    v8f ar, v8f az, v8f axn, v8f ahn, int mbase, int j, int lane,
    const float* bih, const float* bhh, const float* hflt_in,
    float* hflt_out, __bf16* hbf_out, float* det_out, int t)
{
    int n    = j + (lane & 15);
    int mrow = mbase + ((lane >> 4) << 3);
    float br  = bih[n]            + bhh[n];
    float bz  = bih[DET_ + n]     + bhh[DET_ + n];
    float bxn = bih[2 * DET_ + n];
    float bhn = bhh[2 * DET_ + n];
#pragma unroll
    for (int i = 0; i < 8; ++i) {
        int m = mrow + i;
        float r  = sigmoidf_(ar[i] + br);
        float z  = sigmoidf_(az[i] + bz);
        float nn = tanhf((axn[i] + bxn) + r * (ahn[i] + bhn));
        float hp = hflt_in[m * DET_ + n];
        float h  = (1.0f - z) * nn + z * hp;
        hflt_out[m * DET_ + n] = h;
        hbf_out[m * DET_ + n]  = (__bf16)h;
        det_out[(m * T_ + t) * DET_ + n] = h;
    }
}

// Each wave: 2 M-tiles x 1 N-tile of h_new. Gate B-fragments (3 weight matrices)
// are loaded once per K-step and reused across both M-tiles. The 8 waves of a
// block share the same N column -> their B loads hit WGP$.
__global__ void __launch_bounds__(256) k_gru(
    const __bf16* xpad, const __bf16* wihp, const __bf16* hbf_in, const __bf16* whhb,
    const float* bih, const float* bhh, const float* hflt_in,
    float* hflt_out, __bf16* hbf_out, float* det_out, int t)
{
    int lane  = threadIdx.x & 31;
    int wave  = threadIdx.x >> 5;
    int task  = blockIdx.x * 8 + wave;         // 0..511
    int j     = (task >> 4) << 4;              // N column tile (DET)
    int mbase = (task & 15) << 5;              // two row tiles: mbase, mbase+16

    v8f ar0 = {}, az0 = {}, axn0 = {}, ahn0 = {};
    v8f ar1 = {}, az1 = {}, axn1 = {}, ahn1 = {};

    // input-gate contribution, K = 64 (padded)
#pragma unroll
    for (int kb = 0; kb < XPADK; kb += 32) {
        v16bf a0 = load_a_bf16(xpad, XPADK, mbase,      kb, lane);
        v16bf a1 = load_a_bf16(xpad, XPADK, mbase + 16, kb, lane);
        v16bf br = load_b_w(wihp, XPADK, j,            kb, lane);
        v16bf bz = load_b_w(wihp, XPADK, j + DET_,     kb, lane);
        v16bf bn = load_b_w(wihp, XPADK, j + 2 * DET_, kb, lane);
        ar0  = wmma_bf16(a0, br, ar0);   ar1  = wmma_bf16(a1, br, ar1);
        az0  = wmma_bf16(a0, bz, az0);   az1  = wmma_bf16(a1, bz, az1);
        axn0 = wmma_bf16(a0, bn, axn0);  axn1 = wmma_bf16(a1, bn, axn1);
    }
    // hidden-gate contribution, K = 512
#pragma unroll 2
    for (int kb = 0; kb < DET_; kb += 32) {
        v16bf a0 = load_a_bf16(hbf_in, DET_, mbase,      kb, lane);
        v16bf a1 = load_a_bf16(hbf_in, DET_, mbase + 16, kb, lane);
        v16bf br = load_b_w(whhb, DET_, j,            kb, lane);
        v16bf bz = load_b_w(whhb, DET_, j + DET_,     kb, lane);
        v16bf bn = load_b_w(whhb, DET_, j + 2 * DET_, kb, lane);
        ar0  = wmma_bf16(a0, br, ar0);   ar1  = wmma_bf16(a1, br, ar1);
        az0  = wmma_bf16(a0, bz, az0);   az1  = wmma_bf16(a1, bz, az1);
        ahn0 = wmma_bf16(a0, bn, ahn0);  ahn1 = wmma_bf16(a1, bn, ahn1);
    }

    gru_epilogue(ar0, az0, axn0, ahn0, mbase,      j, lane, bih, bhh,
                 hflt_in, hflt_out, hbf_out, det_out, t);
    gru_epilogue(ar1, az1, axn1, ahn1, mbase + 16, j, lane, bih, bhh,
                 hflt_in, hflt_out, hbf_out, det_out, t);
}

// ---------------- k_hidden: trans & rep MLP layer-1 + fused ELU ----------------
// Each wave: 2x2 super-tile -> 4 loads + 4 loads feed 4 WMMAs (2x operand reuse).
__global__ void __launch_bounds__(256) k_hidden(
    const __bf16* hbf, const float* emb, const __bf16* tw1b, const __bf16* rw1b,
    const float* tb1, const float* rb1, __bf16* hidT, __bf16* hidR, int t)
{
    int lane = threadIdx.x & 31;
    int wave = threadIdx.x >> 5;
    int task = blockIdx.x * 8 + wave;          // 0..511
    bool rep = task >= 256;
    int sub  = task & 255;
    int mbase = (sub >> 4) << 5;               // two row tiles
    int nb    = (sub & 15) << 5;               // two col tiles

    const __bf16* w1 = rep ? rw1b : tw1b;
    int ldw = rep ? (DET_ + EMB_) : DET_;

    v8f c00 = {}, c01 = {}, c10 = {}, c11 = {};
#pragma unroll 2
    for (int kb = 0; kb < DET_; kb += 32) {    // h part
        v16bf a0 = load_a_bf16(hbf, DET_, mbase,      kb, lane);
        v16bf a1 = load_a_bf16(hbf, DET_, mbase + 16, kb, lane);
        v16bf b0 = load_b_w(w1, ldw, nb,      kb, lane);
        v16bf b1 = load_b_w(w1, ldw, nb + 16, kb, lane);
        c00 = wmma_bf16(a0, b0, c00);  c01 = wmma_bf16(a0, b1, c01);
        c10 = wmma_bf16(a1, b0, c10);  c11 = wmma_bf16(a1, b1, c11);
    }
    if (rep) {                                 // embedded_obs part, fp32 -> bf16 in regs
        const float* ebase = emb + (size_t)t * EMB_;   // row m at m*(T_*EMB_)
#pragma unroll 2
        for (int kb = 0; kb < EMB_; kb += 32) {
            v16bf a0 = load_a_f32(ebase, T_ * EMB_, mbase,      kb, lane);
            v16bf a1 = load_a_f32(ebase, T_ * EMB_, mbase + 16, kb, lane);
            v16bf b0 = load_b_w(w1, ldw, nb,      DET_ + kb, lane);
            v16bf b1 = load_b_w(w1, ldw, nb + 16, DET_ + kb, lane);
            c00 = wmma_bf16(a0, b0, c00);  c01 = wmma_bf16(a0, b1, c01);
            c10 = wmma_bf16(a1, b0, c10);  c11 = wmma_bf16(a1, b1, c11);
        }
    }

    const float* b1v = rep ? rb1 : tb1;
    __bf16* hid = rep ? hidR : hidT;
    int nc   = lane & 15;
    int mrow = ((lane >> 4) << 3);
    float bn0 = b1v[nb + nc];
    float bn1 = b1v[nb + 16 + nc];
#pragma unroll
    for (int i = 0; i < 8; ++i) {
        int m0 = mbase + mrow + i;
        int m1 = mbase + 16 + mrow + i;
        hid[m0 * HID_ + nb + nc]      = (__bf16)eluf_(c00[i] + bn0);
        hid[m0 * HID_ + nb + 16 + nc] = (__bf16)eluf_(c01[i] + bn1);
        hid[m1 * HID_ + nb + nc]      = (__bf16)eluf_(c10[i] + bn0);
        hid[m1 * HID_ + nb + 16 + nc] = (__bf16)eluf_(c11[i] + bn1);
    }
}

// ---------------- k_tail: layer-2 GEMMs + dist params + rsample + next x ----------------
__global__ void __launch_bounds__(256) k_tail(
    const __bf16* hidT, const __bf16* hidR, const __bf16* tw2b, const __bf16* rw2b,
    const float* tb2, const float* rb2, const float* noise, const float* act,
    float* pm, float* ps, float* qm, float* qs, float* stoch, __bf16* xpad, int t)
{
    int lane = threadIdx.x & 31;
    int wave = threadIdx.x >> 5;
    int task = blockIdx.x * 8 + wave;          // 0..127
    bool rep = task >= 64;
    int sub  = task & 63;
    int mbase = (sub >> 1) << 4;
    int pair  = sub & 1;
    int nb0 = pair * 16;                       // mean columns (0..31)
    int nb1 = STOCH_ + pair * 16;              // raw-std columns (32..63)

    const __bf16* hid = rep ? hidR : hidT;
    const __bf16* w2  = rep ? rw2b : tw2b;
    const float*  b2  = rep ? rb2  : tb2;

    v8f am = {}, as = {};
#pragma unroll 4
    for (int kb = 0; kb < HID_; kb += 32) {
        v16bf a = load_a_bf16(hid, HID_, mbase, kb, lane);
        am = wmma_bf16(a, load_b_w(w2, HID_, nb0, kb, lane), am);
        as = wmma_bf16(a, load_b_w(w2, HID_, nb1, kb, lane), as);
    }

    int n    = nb0 + (lane & 15);              // stoch index 0..31
    int mrow = mbase + ((lane >> 4) << 3);
    float bm = b2[n];
    float bs = b2[STOCH_ + n];
#pragma unroll
    for (int i = 0; i < 8; ++i) {
        int m = mrow + i;
        float mean = am[i] + bm;
        float std  = softplusf_(as[i] + bs) + MIN_STD;
        int o = (m * T_ + t) * STOCH_ + n;
        if (!rep) {
            pm[o] = mean;
            ps[o] = std;
        } else {
            qm[o] = mean;
            qs[o] = std;
            float sv = mean + std * noise[o];
            stoch[o] = sv;
            xpad[m * XPADK + n] = (__bf16)sv;                       // s part of next x
            if (n < ACT_ && (t + 1) < T_)                           // a_{t+1} part
                xpad[m * XPADK + STOCH_ + n] = (__bf16)act[(m * T_ + (t + 1)) * ACT_ + n];
        }
    }
}

// ---------------- host side ----------------
extern "C" void kernel_launch(void* const* d_in, const int* in_sizes, int n_in,
                              void* d_out, int out_size, void* d_ws, size_t ws_size,
                              hipStream_t stream) {
    const float* emb   = (const float*)d_in[0];
    const float* act   = (const float*)d_in[1];
    const float* noise = (const float*)d_in[2];
    const float* wih   = (const float*)d_in[3];
    const float* whh   = (const float*)d_in[4];
    const float* bih   = (const float*)d_in[5];
    const float* bhh   = (const float*)d_in[6];
    const float* tw1   = (const float*)d_in[7];
    const float* tb1   = (const float*)d_in[8];
    const float* tw2   = (const float*)d_in[9];
    const float* tb2   = (const float*)d_in[10];
    const float* rw1   = (const float*)d_in[11];
    const float* rb1   = (const float*)d_in[12];
    const float* rw2   = (const float*)d_in[13];
    const float* rb2   = (const float*)d_in[14];

    float* out   = (float*)d_out;                      // concatenated tuple
    float* det   = out;                                // [B,T,DET]
    float* stoch = out + (size_t)B_ * T_ * DET_;       // [B,T,STOCH]
    float* pm    = stoch + (size_t)B_ * T_ * STOCH_;
    float* ps    = pm    + (size_t)B_ * T_ * STOCH_;
    float* qm    = ps    + (size_t)B_ * T_ * STOCH_;
    float* qs    = qm    + (size_t)B_ * T_ * STOCH_;

    char* w = (char*)d_ws;
    size_t off = 0;
    __bf16* wihp  = (__bf16*)(w + off); off += (size_t)SZ_WIHP * 2;
    __bf16* whhb  = (__bf16*)(w + off); off += (size_t)SZ_WHH  * 2;
    __bf16* tw1b  = (__bf16*)(w + off); off += (size_t)SZ_TW1  * 2;
    __bf16* tw2b  = (__bf16*)(w + off); off += (size_t)SZ_TW2  * 2;
    __bf16* rw1b  = (__bf16*)(w + off); off += (size_t)SZ_RW1  * 2;
    __bf16* rw2b  = (__bf16*)(w + off); off += (size_t)SZ_RW2  * 2;
    __bf16* hbf0  = (__bf16*)(w + off); off += (size_t)SZ_H * 2;
    __bf16* hbf1  = (__bf16*)(w + off); off += (size_t)SZ_H * 2;
    float*  hflt0 = (float*)(w + off);  off += (size_t)SZ_H * 4;
    float*  hflt1 = (float*)(w + off);  off += (size_t)SZ_H * 4;
    __bf16* xpad  = (__bf16*)(w + off); off += (size_t)SZ_XPAD * 2;
    __bf16* hidT  = (__bf16*)(w + off); off += (size_t)B_ * HID_ * 2;
    __bf16* hidR  = (__bf16*)(w + off); off += (size_t)B_ * HID_ * 2;

    k_prep<<<2048, 256, 0, stream>>>(wih, whh, tw1, tw2, rw1, rw2, act,
                                     wihp, whhb, tw1b, tw2b, rw1b, rw2b,
                                     hbf0, hflt0, xpad);

    __bf16* hbf[2]  = {hbf0, hbf1};
    float*  hflt[2] = {hflt0, hflt1};
    for (int t = 0; t < T_; ++t) {
        int p = t & 1;
        k_gru<<<64, 256, 0, stream>>>(xpad, wihp, hbf[p], whhb, bih, bhh,
                                      hflt[p], hflt[1 - p], hbf[1 - p], det, t);
        k_hidden<<<64, 256, 0, stream>>>(hbf[1 - p], emb, tw1b, rw1b, tb1, rb1,
                                         hidT, hidR, t);
        k_tail<<<16, 256, 0, stream>>>(hidT, hidR, tw2b, rw2b, tb2, rb2, noise, act,
                                       pm, ps, qm, qs, stoch, xpad, t);
    }
}
